// Model_63556926046481
// MI455X (gfx1250) — matrix operand order repaired and checked by reading, not compiled, not run on hardware
//
#include <hip/hip_runtime.h>
#include <hip/hip_bf16.h>

// ---------------------------------------------------------------------------
// Types for WMMA fragments (CDNA5 / gfx1250, wave32)
// ---------------------------------------------------------------------------
typedef __attribute__((ext_vector_type(16))) __bf16 v16bf;
typedef __attribute__((ext_vector_type(8)))  float  v8f;
typedef __attribute__((ext_vector_type(4)))  unsigned int u32x4;

union FragAB {
    unsigned int u[8];
    v16bf v;
};

// round-to-nearest-even f32 -> bf16 (stored as ushort)
static __device__ __forceinline__ unsigned short f2bf(float f) {
    unsigned int u = __float_as_uint(f);
    unsigned int r = (u + 0x7FFFu + ((u >> 16) & 1u)) >> 16;
    return (unsigned short)r;
}

// ---------------------------------------------------------------------------
// f32 -> bf16 conversion
// ---------------------------------------------------------------------------
__global__ void cvt_bf16(const float* __restrict__ in, unsigned short* __restrict__ out, size_t n) {
    size_t i = (size_t)blockIdx.x * blockDim.x + threadIdx.x;
    if (i < n) out[i] = f2bf(in[i]);
}

// ---------------------------------------------------------------------------
// RMSNorm (D=512) producing bf16 activations
// ---------------------------------------------------------------------------
__global__ __launch_bounds__(256) void rmsnorm_bf16(const float* __restrict__ x,
                                                    const float* __restrict__ w,
                                                    unsigned short* __restrict__ o, int D) {
    int row = blockIdx.x;
    const float* xr = x + (size_t)row * D;
    float ss = 0.f;
    for (int i = threadIdx.x; i < D; i += 256) { float v = xr[i]; ss += v * v; }
    __shared__ float red[256];
    red[threadIdx.x] = ss; __syncthreads();
    for (int s = 128; s > 0; s >>= 1) {
        if (threadIdx.x < s) red[threadIdx.x] += red[threadIdx.x + s];
        __syncthreads();
    }
    float inv = rsqrtf(red[0] / (float)D + 1e-6f);
    for (int i = threadIdx.x; i < D; i += 256)
        o[(size_t)row * D + i] = f2bf(xr[i] * inv * w[i]);
}

// ---------------------------------------------------------------------------
// bf16 WMMA GEMM, register-blocked: C[M,N] = A[M,K] * W[N,K]^T (+ residual).
// Each wave computes a 64x64 C block = 4x4 WMMA tiles (16 f32x8 accumulators).
// Per K=32 step: 4 A-fragments + 4 B-fragments (16 b128 loads) feed 16
// independent v_wmma_f32_16x16x32_bf16 ops -> ~4x the arithmetic intensity of
// a 1-tile wave, keeping the matrix pipe (not L2) as the limiter.
// Block = 8 waves arranged 2(M) x 4(N) -> 128x256 block tile.
// Fragment layouts follow CDNA5 ISA 7.12.2 (16-bit A 16x32, B 32x16).
// ---------------------------------------------------------------------------
__global__ __launch_bounds__(256) void gemm_bf16(const unsigned short* __restrict__ A,
                                                 const unsigned short* __restrict__ W,
                                                 const float* __restrict__ resid,
                                                 float* __restrict__ C,
                                                 int M, int N, int K) {
    int lane = threadIdx.x & 31;
    int wave = threadIdx.x >> 5;
    int wm = wave >> 2;                       // 0..1
    int wn = wave & 3;                        // 0..3
    int mbase = blockIdx.y * 128 + wm * 64;   // wave's 64-row slab
    int nbase = blockIdx.x * 256 + wn * 64;   // wave's 64-col slab
    if (mbase >= M || nbase >= N) return;     // wave-uniform: EXEC all-ones at WMMA

    int half = lane >> 4;                     // 0: lanes 0-15, 1: lanes 16-31
    int l15  = lane & 15;
    int kA   = half ? 8 : 0;                  // A frag K base within 32-step
    int kB   = half ? 8 : 0;                  // B frag K base within 32-step

    const unsigned short* arow[4];
    const unsigned short* wrow[4];
#pragma unroll
    for (int i = 0; i < 4; i++) {
        arow[i] = A + (size_t)(mbase + i * 16 + l15) * K;
        wrow[i] = W + (size_t)(nbase + i * 16 + l15) * K;
    }

    v8f acc[4][4] = {};
    for (int k0 = 0; k0 < K; k0 += 32) {
        FragAB a[4], b[4];
#pragma unroll
        for (int i = 0; i < 4; i++) {
            // A lane: K = k0+kA..+7 (VGPR0-3) and k0+kA+16..+23 (VGPR4-7)
            *(u32x4*)&a[i].u[0] = *(const u32x4*)(arow[i] + k0 + kA);
            *(u32x4*)&a[i].u[4] = *(const u32x4*)(arow[i] + k0 + kA + 16);
            // B lane: K = k0+kB..+7 (VGPR0-3) and k0+kB+16..+23 (VGPR4-7)
            *(u32x4*)&b[i].u[0] = *(const u32x4*)(wrow[i] + k0 + kB);
            *(u32x4*)&b[i].u[4] = *(const u32x4*)(wrow[i] + k0 + kB + 16);
        }
#pragma unroll
        for (int mi = 0; mi < 4; mi++)
#pragma unroll
            for (int ni = 0; ni < 4; ni++)
                acc[mi][ni] = __builtin_amdgcn_wmma_f32_16x16x32_bf16(
                    false, a[mi].v, false, b[ni].v, (short)0, acc[mi][ni], false, false);
    }

    // C/D layout: lane n = lane&15; VGPR v holds M = v + (half?8:0)
#pragma unroll
    for (int mi = 0; mi < 4; mi++) {
        int mb = mbase + mi * 16 + (half ? 8 : 0);
#pragma unroll
        for (int ni = 0; ni < 4; ni++) {
            int nc = nbase + ni * 16 + l15;
#pragma unroll
            for (int v = 0; v < 8; v++) {
                size_t offn = (size_t)(mb + v) * N + nc;
                float  val  = acc[mi][ni][v];
                if (resid) val += resid[offn];
                C[offn] = val;
            }
        }
    }
}

// ---------------------------------------------------------------------------
// RoPE + head split: qkv[M,1536] -> qh/kh/vh [(g*8+h)*Sa + r][64]
// pos = (g % Lmod)*Sa + r  (encoder),  pos = r (router, Lmod=0)
// ---------------------------------------------------------------------------
__global__ __launch_bounds__(256) void rope_split(const float* __restrict__ qkv,
                                                  float* __restrict__ qh,
                                                  float* __restrict__ kh,
                                                  float* __restrict__ vh,
                                                  int Sa, int Lmod) {
    int g = blockIdx.y, r = blockIdx.x;
    int row = g * Sa + r;
    int pos = Lmod ? ((g % Lmod) * Sa + r) : r;
    const float* src = qkv + (size_t)row * 1536;
    for (int i = threadIdx.x; i < 512; i += 256) {
        int h = i >> 6, a = i & 63;
        int ai = a & 31; bool hi = a >= 32;
        float freq = (float)pos * __powf(10000.0f, -(float)(2 * ai) / 64.0f);
        float c = __cosf(freq), s = __sinf(freq);
        float q1 = src[h * 64 + ai],        q2 = src[h * 64 + 32 + ai];
        float k1 = src[512 + h * 64 + ai],  k2 = src[512 + h * 64 + 32 + ai];
        float qr = hi ? (-q1 * s + q2 * c) : (q1 * c + q2 * s);
        float kr = hi ? (-k1 * s + k2 * c) : (k1 * c + k2 * s);
        size_t dst = (((size_t)(g * 8 + h)) * Sa + r) * 64 + a;
        qh[dst] = qr; kh[dst] = kr; vh[dst] = src[1024 + i];
    }
}

// ---------------------------------------------------------------------------
// Attention (online softmax). One block per (group, head); thread t = query row t.
// K/V tiles of 64 rows staged through LDS (32KB). Output written token-major
// into attn_out[token][h*64 + c].
// ---------------------------------------------------------------------------
__global__ __launch_bounds__(256) void attn_kernel(const float* __restrict__ qh,
                                                   const float* __restrict__ kh,
                                                   const float* __restrict__ vh,
                                                   float* __restrict__ xout, int Sa) {
    __shared__ float sK[64 * 64];
    __shared__ float sV[64 * 64];
    int gh = blockIdx.x;
    int g = gh >> 3, h = gh & 7;
    const float* Q = qh + (size_t)gh * Sa * 64;
    const float* K = kh + (size_t)gh * Sa * 64;
    const float* V = vh + (size_t)gh * Sa * 64;
    int t = threadIdx.x;
    bool act = t < Sa;

    float q[64], acc[64];
    float m = -1e30f, l = 0.f;
    if (act) {
#pragma unroll
        for (int i = 0; i < 64; i++) { q[i] = Q[t * 64 + i]; acc[i] = 0.f; }
    }
    const float scale = 0.125f;  // 1/sqrt(64)

    for (int j0 = 0; j0 < Sa; j0 += 64) {
        int jn = (Sa - j0 < 64) ? (Sa - j0) : 64;
        __syncthreads();
        for (int i = t; i < jn * 64; i += 256) { sK[i] = K[j0 * 64 + i]; sV[i] = V[j0 * 64 + i]; }
        __syncthreads();
        if (act) {
            for (int j = 0; j < jn; j++) {
                float s = 0.f;
#pragma unroll
                for (int i = 0; i < 64; i++) s += q[i] * sK[j * 64 + i];
                s *= scale;
                float mn = fmaxf(m, s);
                float corr = __expf(m - mn);
                float p = __expf(s - mn);
                l = l * corr + p;
#pragma unroll
                for (int i = 0; i < 64; i++) acc[i] = acc[i] * corr + p * sV[j * 64 + i];
                m = mn;
            }
        }
    }
    if (act) {
        float inv = 1.f / l;
        size_t row = (size_t)g * Sa + t;
#pragma unroll
        for (int i = 0; i < 64; i++) xout[row * 512 + h * 64 + i] = acc[i] * inv;
    }
}

// ---------------------------------------------------------------------------
// silu(x1) * x2 -> bf16 gated activations; up[M,2048] split 1024/1024
// ---------------------------------------------------------------------------
__global__ void silu_gate(const float* __restrict__ up, unsigned short* __restrict__ g, size_t M) {
    size_t idx = (size_t)blockIdx.x * blockDim.x + threadIdx.x;
    if (idx >= M * 1024) return;
    size_t r = idx >> 10; int c = (int)(idx & 1023);
    float a = up[r * 2048 + c], b = up[r * 2048 + 1024 + c];
    float s = a / (1.f + __expf(-a));
    g[idx] = f2bf(s * b);
}

// ---------------------------------------------------------------------------
// Build xc: 128 sequences of 136 rows (128 encoded tokens + 8 router tokens)
// ---------------------------------------------------------------------------
__global__ void build_xc(const float* __restrict__ enc, const float* __restrict__ rtok,
                         float* __restrict__ xc) {
    int j = blockIdx.y, r = blockIdx.x;  // r in [0,136)
    const float* src = (r < 128) ? enc + ((size_t)j * 128 + r) * 512
                                 : rtok + (size_t)(r - 128) * 512;
    float* dst = xc + ((size_t)j * 136 + r) * 512;
    for (int i = threadIdx.x; i < 512; i += 256) dst[i] = src[i];
}

__global__ void gather_rt(const float* __restrict__ xc, float* __restrict__ o) {
    int j = blockIdx.y, r = blockIdx.x;  // r in [0,8)
    const float* s = xc + ((size_t)j * 136 + 128 + r) * 512;
    float* d = o + ((size_t)j * 8 + r) * 512;
    for (int i = threadIdx.x; i < 512; i += 256) d[i] = s[i];
}

// ---------------------------------------------------------------------------
// Router norms: out[1024,256] -> xr_n[ln][j][128], xg_n[ln][j][128]
// ---------------------------------------------------------------------------
__global__ __launch_bounds__(128) void router_norm(const float* __restrict__ outb,
                                                   const float* __restrict__ rw,
                                                   const float* __restrict__ gw,
                                                   float* __restrict__ xr, float* __restrict__ xg) {
    int ln = blockIdx.x >> 7, j = blockIdx.x & 127;
    const float* row = outb + ((size_t)j * 8 + ln) * 256;
    __shared__ float red[128];
    int t = threadIdx.x;
    float vr = row[t], vg = row[128 + t];
    red[t] = vr * vr; __syncthreads();
    for (int s = 64; s > 0; s >>= 1) { if (t < s) red[t] += red[t + s]; __syncthreads(); }
    float invr = rsqrtf(red[0] / 128.f + 1e-6f); __syncthreads();
    red[t] = vg * vg; __syncthreads();
    for (int s = 64; s > 0; s >>= 1) { if (t < s) red[t] += red[t + s]; __syncthreads(); }
    float invg = rsqrtf(red[0] / 128.f + 1e-6f);
    size_t o = ((size_t)ln * 128 + j) * 128 + t;
    xr[o] = vr * invr * rw[t];
    xg[o] = vg * invg * gw[t];
}

// logits[k][j][e] = dot(xn[k>>2][j], keys[k][:,e]);  32*128*16 threads
__global__ void logits_k(const float* __restrict__ xn, const float* __restrict__ keys,
                         float* __restrict__ lg) {
    int idx = blockIdx.x * 256 + threadIdx.x;
    if (idx >= 32 * 128 * 16) return;
    int e = idx & 15, j = (idx >> 4) & 127, k = idx >> 11;
    int ln = k >> 2;
    const float* xv = xn + ((size_t)ln * 128 + j) * 128;
    const float* kv = keys + (size_t)k * 128 * 16 + e;
    float s = 0.f;
    for (int d = 0; d < 128; d++) s += xv[d] * kv[d * 16];
    lg[((size_t)k * 128 + j) * 16 + e] = s;
}

// top-2 over 16 router logits; pack indices (+gn*16) and rv+gv values
__global__ void topk_pack(const float* __restrict__ rl, const float* __restrict__ gl,
                          float* __restrict__ out_idx, float* __restrict__ out_val) {
    int idx = blockIdx.x * 256 + threadIdx.x;
    if (idx >= 32 * 128) return;
    int j = idx & 127, k = idx >> 7;
    int ln = k >> 2, gn = k & 3;
    const float* r = rl + (size_t)idx * 16;
    const float* g = gl + (size_t)idx * 16;
    int i0 = 0; float v0 = r[0];
    for (int e = 1; e < 16; e++) if (r[e] > v0) { v0 = r[e]; i0 = e; }
    int i1 = (i0 == 0) ? 1 : 0; float v1 = r[i1];
    for (int e = 0; e < 16; e++) { if (e == i0 || e == i1) continue; if (r[e] > v1) { v1 = r[e]; i1 = e; } }
    int base = ln * 1024 + j * 8 + gn * 2;
    out_idx[base + 0] = (float)(i0 + gn * 16);
    out_idx[base + 1] = (float)(i1 + gn * 16);
    out_val[base + 0] = v0 + g[i0];
    out_val[base + 1] = v1 + g[i1];
}

// balance loss: single block
__global__ __launch_bounds__(256) void balance(const float* __restrict__ rl,
                                               const float* __restrict__ out_idx,
                                               float* __restrict__ loss) {
    __shared__ float hist[64];
    __shared__ float vb[64];
    __shared__ float red[256];
    float total = 0.f;
    int t = threadIdx.x;
    for (int ln = 0; ln < 8; ln++) {
        if (t < 64) { hist[t] = 0.f; vb[t] = 0.f; }
        __syncthreads();
        for (int i = t; i < 1024; i += 256) {
            int e = (int)out_idx[ln * 1024 + i];
            atomicAdd(&hist[e], 1.0f / 1024.f);
        }
        if (t < 128) {
            int j = t;
            float va[64]; float mx = -1e30f;
            for (int gn = 0; gn < 4; gn++)
                for (int e = 0; e < 16; e++) {
                    float v = rl[(((size_t)(ln * 4 + gn)) * 128 + j) * 16 + e];
                    va[gn * 16 + e] = v; mx = fmaxf(mx, v);
                }
            float s = 0.f;
            for (int q = 0; q < 64; q++) { va[q] = __expf(va[q] - mx); s += va[q]; }
            float inv = 1.f / (s * 128.f);
            for (int q = 0; q < 64; q++) atomicAdd(&vb[q], va[q] * inv);
        }
        __syncthreads();
        if (t < 64) total += hist[t] * vb[t];
        __syncthreads();
    }
    red[t] = total; __syncthreads();
    for (int s = 128; s > 0; s >>= 1) { if (t < s) red[t] += red[t + s]; __syncthreads(); }
    if (t == 0) loss[0] = red[0] * 64.f * 0.01f;
}

// ---------------------------------------------------------------------------
// Host orchestration
// ---------------------------------------------------------------------------
struct LayerW {
    const unsigned short *qkv, *o, *up, *dn;
    const float *anw, *fnw;
};

static void run_layer(float* X, const LayerW& w, int nG, int Sa, int Lmod,
                      unsigned short* hnb, float* big0,
                      float* qh, float* kh, float* vh, float* attn_out,
                      unsigned short* gated, hipStream_t stream) {
    int M = nG * Sa;
    int mg = M / 128;  // GEMM grid.y (all M here are multiples of 128)
    rmsnorm_bf16<<<M, 256, 0, stream>>>(X, w.anw, hnb, 512);
    gemm_bf16<<<dim3(1536 / 256, mg), 256, 0, stream>>>(hnb, w.qkv, nullptr, big0, M, 1536, 512);
    rope_split<<<dim3(Sa, nG), 256, 0, stream>>>(big0, qh, kh, vh, Sa, Lmod);
    attn_kernel<<<nG * 8, 256, 0, stream>>>(qh, kh, vh, attn_out, Sa);
    cvt_bf16<<<(unsigned)(((size_t)M * 512 + 255) / 256), 256, 0, stream>>>(attn_out, hnb, (size_t)M * 512);
    gemm_bf16<<<dim3(512 / 256, mg), 256, 0, stream>>>(hnb, w.o, X, X, M, 512, 512);
    rmsnorm_bf16<<<M, 256, 0, stream>>>(X, w.fnw, hnb, 512);
    gemm_bf16<<<dim3(2048 / 256, mg), 256, 0, stream>>>(hnb, w.up, nullptr, big0, M, 2048, 512);
    silu_gate<<<(unsigned)(((size_t)M * 1024 + 255) / 256), 256, 0, stream>>>(big0, gated, (size_t)M);
    gemm_bf16<<<dim3(512 / 256, mg), 256, 0, stream>>>(gated, w.dn, X, X, M, 512, 1024);
}

extern "C" void kernel_launch(void* const* d_in, const int* in_sizes, int n_in,
                              void* d_out, int out_size, void* d_ws, size_t ws_size,
                              hipStream_t stream) {
    const float* x            = (const float*)d_in[0];
    const float* enc_attn_w   = (const float*)d_in[1];
    const float* enc_attno_w  = (const float*)d_in[2];
    const float* enc_up_w     = (const float*)d_in[3];
    const float* enc_down_w   = (const float*)d_in[4];
    const float* enc_anorm_w  = (const float*)d_in[5];
    const float* enc_fnorm_w  = (const float*)d_in[6];
    const float* rt_attn_w    = (const float*)d_in[7];
    const float* rt_attno_w   = (const float*)d_in[8];
    const float* rt_up_w      = (const float*)d_in[9];
    const float* rt_down_w    = (const float*)d_in[10];
    const float* rt_anorm_w   = (const float*)d_in[11];
    const float* rt_fnorm_w   = (const float*)d_in[12];
    const float* router_token = (const float*)d_in[13];
    const float* out_w        = (const float*)d_in[14];
    const float* keys_router  = (const float*)d_in[15];
    const float* keys_gate    = (const float*)d_in[16];
    const float* rnorm_w      = (const float*)d_in[17];
    const float* gnorm_w      = (const float*)d_in[18];
    float* out = (float*)d_out;

    const size_t M_ENC = 16384, M_RT = 17408;

    char* base = (char*)d_ws;
    size_t off = 0;
    auto carve = [&](size_t bytes) -> void* {
        void* p = base + off;
        off += (bytes + 255) & ~(size_t)255;
        return p;
    };

    unsigned short* w_eattn  = (unsigned short*)carve((size_t)2 * 1536 * 512 * 2);
    unsigned short* w_eattno = (unsigned short*)carve((size_t)2 * 512 * 512 * 2);
    unsigned short* w_eup    = (unsigned short*)carve((size_t)2 * 2048 * 512 * 2);
    unsigned short* w_edn    = (unsigned short*)carve((size_t)2 * 512 * 1024 * 2);
    unsigned short* w_rattn  = (unsigned short*)carve((size_t)2 * 1536 * 512 * 2);
    unsigned short* w_rattno = (unsigned short*)carve((size_t)2 * 512 * 512 * 2);
    unsigned short* w_rup    = (unsigned short*)carve((size_t)2 * 2048 * 512 * 2);
    unsigned short* w_rdn    = (unsigned short*)carve((size_t)2 * 512 * 1024 * 2);
    unsigned short* w_out    = (unsigned short*)carve((size_t)256 * 512 * 2);

    float*          xres   = (float*)carve(M_ENC * 512 * 4);
    float*          xc     = (float*)carve(M_RT * 512 * 4);
    unsigned short* hnb    = (unsigned short*)carve(M_RT * 512 * 2);
    float*          big0   = (float*)carve(M_RT * 2048 * 4);
    float*          qh     = (float*)carve(M_RT * 512 * 4);
    float*          kh     = (float*)carve(M_RT * 512 * 4);
    float*          vh     = (float*)carve(M_RT * 512 * 4);
    float*          attn_o = (float*)carve(M_RT * 512 * 4);
    unsigned short* gated  = (unsigned short*)carve(M_RT * 1024 * 2);
    float*          rtin   = (float*)carve((size_t)1024 * 512 * 4);
    float*          outbuf = (float*)carve((size_t)1024 * 256 * 4);
    float*          xrn    = (float*)carve((size_t)131072 * 4);
    float*          xgn    = (float*)carve((size_t)131072 * 4);
    float*          rlog   = (float*)carve((size_t)65536 * 4);
    float*          glog   = (float*)carve((size_t)65536 * 4);

    auto cv = [&](const float* s, unsigned short* d, size_t n) {
        cvt_bf16<<<(unsigned)((n + 255) / 256), 256, 0, stream>>>(s, d, n);
    };
    cv(enc_attn_w,  w_eattn,  (size_t)2 * 1536 * 512);
    cv(enc_attno_w, w_eattno, (size_t)2 * 512 * 512);
    cv(enc_up_w,    w_eup,    (size_t)2 * 2048 * 512);
    cv(enc_down_w,  w_edn,    (size_t)2 * 512 * 1024);
    cv(rt_attn_w,   w_rattn,  (size_t)2 * 1536 * 512);
    cv(rt_attno_w,  w_rattno, (size_t)2 * 512 * 512);
    cv(rt_up_w,     w_rup,    (size_t)2 * 2048 * 512);
    cv(rt_down_w,   w_rdn,    (size_t)2 * 512 * 1024);
    cv(out_w,       w_out,    (size_t)256 * 512);

    hipMemcpyAsync(xres, x, M_ENC * 512 * 4, hipMemcpyDeviceToDevice, stream);

    // encoder layers: 128 groups (b,l) of 128 tokens, rope pos = l*128 + r
    for (int i = 0; i < 2; i++) {
        LayerW w = { w_eattn  + (size_t)i * 1536 * 512,
                     w_eattno + (size_t)i * 512 * 512,
                     w_eup    + (size_t)i * 2048 * 512,
                     w_edn    + (size_t)i * 512 * 1024,
                     enc_anorm_w + (size_t)i * 512,
                     enc_fnorm_w + (size_t)i * 512 };
        run_layer(xres, w, 128, 128, 32, hnb, big0, qh, kh, vh, attn_o, gated, stream);
    }

    // concat router tokens -> xc (128 x 136 x 512)
    build_xc<<<dim3(136, 128), 256, 0, stream>>>(xres, router_token, xc);

    // router layers: 128 groups of 136 tokens, rope pos = r
    for (int i = 0; i < 2; i++) {
        LayerW w = { w_rattn  + (size_t)i * 1536 * 512,
                     w_rattno + (size_t)i * 512 * 512,
                     w_rup    + (size_t)i * 2048 * 512,
                     w_rdn    + (size_t)i * 512 * 1024,
                     rt_anorm_w + (size_t)i * 512,
                     rt_fnorm_w + (size_t)i * 512 };
        run_layer(xc, w, 128, 136, 0, hnb, big0, qh, kh, vh, attn_o, gated, stream);
    }

    // out = xc[:,128:] @ out_w^T  -> (1024, 256)
    gather_rt<<<dim3(8, 128), 256, 0, stream>>>(xc, rtin);
    cvt_bf16<<<(unsigned)(((size_t)1024 * 512 + 255) / 256), 256, 0, stream>>>(rtin, hnb, (size_t)1024 * 512);
    gemm_bf16<<<dim3(1, 1024 / 128), 256, 0, stream>>>(hnb, w_out, nullptr, outbuf, 1024, 256, 512);

    // router head
    router_norm<<<1024, 128, 0, stream>>>(outbuf, rnorm_w, gnorm_w, xrn, xgn);
    logits_k<<<(32 * 128 * 16) / 256, 256, 0, stream>>>(xrn, keys_router, rlog);
    logits_k<<<(32 * 128 * 16) / 256, 256, 0, stream>>>(xgn, keys_gate, glog);

    // d_out: indices [0,8192) | values [8192,16384) | loss [16384]
    topk_pack<<<16, 256, 0, stream>>>(rlog, glog, out, out + 8192);
    balance<<<1, 256, 0, stream>>>(rlog, out, out + 16384);

    (void)in_sizes; (void)n_in; (void)out_size; (void)ws_size;
}
